// EntityEncoder_34024730919229
// MI455X (gfx1250) — matrix operand order repaired and checked by reading, not compiled, not run on hardware
//
#include <hip/hip_runtime.h>

#define B_  32
#define S_  256
#define H_  1024
#define HL_ 512
#define G4_ 2048   // 4*HL
#define E_  256
#define L_  5

#define KC_ 128                 // K chunk staged in LDS
#define BPAD_ 8                 // pad to break LDS bank conflicts
#define BSTRIDE_ (KC_ + BPAD_)  // 136 halfwords per staged B row (272 B)

#if defined(__has_builtin)
#if __has_builtin(__builtin_amdgcn_global_load_async_to_lds_b128)
#define HAVE_ASYNC_LDS 1
#endif
#endif
#ifndef HAVE_ASYNC_LDS
#define HAVE_ASYNC_LDS 0
#endif

typedef __attribute__((ext_vector_type(16))) __bf16 v16bf;
typedef __attribute__((ext_vector_type(8)))  float  v8f;

// Exact pointee type the async builtin expects (per hipcc diagnostic):
// "__attribute__((__vector_size__(4 * sizeof(int)))) int __device__ *"
typedef int async_v4i __attribute__((__vector_size__(4 * sizeof(int))));
typedef __attribute__((address_space(1))) async_v4i* g_v4i_ptr;
typedef __attribute__((address_space(3))) async_v4i* l_v4i_ptr;

union BF16Frag {
  v16bf v;
  uint4 q[2];
};

__device__ __forceinline__ unsigned short f2bf(float f) {
  unsigned int x = __float_as_uint(f);
  unsigned int r = x + 0x7FFFu + ((x >> 16) & 1u);   // round-to-nearest-even
  return (unsigned short)(r >> 16);
}
__device__ __forceinline__ float bf2f(unsigned short h) {
  return __uint_as_float(((unsigned int)h) << 16);
}
__device__ __forceinline__ float sigm_(float x) {
  return 1.0f / (1.0f + __expf(-x));
}

__device__ __forceinline__ v8f wmma_bf16(const BF16Frag& a, const BF16Frag& b, v8f acc) {
  return __builtin_amdgcn_wmma_f32_16x16x32_bf16(false, a.v, false, b.v,
                                                 (short)0, acc, false, false);
}

// Wait for all of this wave's async LDS loads, then workgroup barrier.
__device__ __forceinline__ void async_barrier() {
#if HAVE_ASYNC_LDS
#if __has_builtin(__builtin_amdgcn_s_wait_asynccnt)
  __builtin_amdgcn_s_wait_asynccnt(0);
#else
  asm volatile("s_wait_asynccnt 0" ::: "memory");
#endif
#endif
  __syncthreads();
}

// ---------------------------------------------------------------------------
// Stage one K-chunk (KC_=128) of a 64-column B panel (Bt stored [N,K]) into
// LDS.  64 rows x 128 halfwords = 1024 x 16B transfers; 256 threads x 4.
// Async (ASYNCcnt-tracked) when the toolchain exposes the builtin.
// ---------------------------------------------------------------------------
__device__ __forceinline__ void stage_B(const unsigned short* Bt, int ldb,
                                        int panelCol0, int kb,
                                        unsigned short* buf, int tid) {
  for (int t = 0; t < 4; ++t) {
    int i = tid + t * 256;
    int row = i >> 4;           // 0..63  (panel column)
    int piece = i & 15;         // 16B piece within the 256B row chunk
    const unsigned short* src =
        Bt + (size_t)(panelCol0 + row) * ldb + kb + piece * 8;
    unsigned short* dst = buf + row * BSTRIDE_ + piece * 8;
#if HAVE_ASYNC_LDS
    g_v4i_ptr gsrc = (g_v4i_ptr)(__attribute__((address_space(1))) void*)(void*)src;
    l_v4i_ptr ldst = (l_v4i_ptr)(__attribute__((address_space(3))) void*)(void*)dst;
    __builtin_amdgcn_global_load_async_to_lds_b128(gsrc, ldst, 0, 0);
#else
    *(uint4*)dst = *(const uint4*)src;
#endif
  }
}

// ---------------------------------------------------------------------------
// Blocked GEMM core: block = 8 waves stacked on M (256 rows) x 64-col panel.
// Each wave: 2x4 grid of 16x16 acc tiles.  A fragments from global, B
// fragments from the double-buffered LDS panel.
// CDNA5 16-bit A layout: lane (l&15)=row, K runs of 8 at kb+8*(l>>4) and +16.
// B layout: lane (l&15)=col, K runs of 8 at kb+8*(l>>4) and +16 (same as A).
// ---------------------------------------------------------------------------
__device__ __forceinline__ void gemm_panel(const unsigned short* Abase, int lda,
                                           const unsigned short* Bt, int ldb,
                                           int K, int panelCol0,
                                           unsigned short* smem,
                                           v8f acc[2][4]) {
  int tid = threadIdx.x;
  int w = tid >> 5, lane = tid & 31;
  int r = lane & 15, half = lane >> 4;
  const unsigned short* arow0 = Abase + (size_t)(w * 32 + r) * lda;
  const unsigned short* arow1 = arow0 + (size_t)16 * lda;

  stage_B(Bt, ldb, panelCol0, 0, smem, tid);
  async_barrier();

  for (int kb = 0; kb < K; kb += KC_) {
    int p = (kb / KC_) & 1;
    unsigned short* cur = smem + p * (64 * BSTRIDE_);
    unsigned short* nxt = smem + (p ^ 1) * (64 * BSTRIDE_);
    if (kb + KC_ < K) stage_B(Bt, ldb, panelCol0, kb + KC_, nxt, tid);

    for (int kk = 0; kk < KC_; kk += 32) {
      BF16Frag a0, a1;
      a0.q[0] = *(const uint4*)(arow0 + kb + kk + half * 8);
      a0.q[1] = *(const uint4*)(arow0 + kb + kk + half * 8 + 16);
      a1.q[0] = *(const uint4*)(arow1 + kb + kk + half * 8);
      a1.q[1] = *(const uint4*)(arow1 + kb + kk + half * 8 + 16);
      for (int n = 0; n < 4; ++n) {
        BF16Frag b;
        const unsigned short* bp = cur + (n * 16 + r) * BSTRIDE_ + kk + half * 8;
        b.q[0] = *(const uint4*)bp;
        b.q[1] = *(const uint4*)(bp + 16);
        acc[0][n] = wmma_bf16(a0, b, acc[0][n]);
        acc[1][n] = wmma_bf16(a1, b, acc[1][n]);
      }
    }
    async_barrier();   // chunk kb+KC_ landed; all reads of `nxt`'s old data done
  }
}

// Small non-blocked tile loop (kept for the LSTM recurrence: A from LDS).
__device__ __forceinline__ v8f wmma_loop(const unsigned short* arow,
                                         const unsigned short* brow,
                                         int K, int half, v8f acc) {
  for (int kb = 0; kb < K; kb += 32) {
    BF16Frag a, b;
    a.q[0] = *(const uint4*)(arow + kb + half * 8);
    a.q[1] = *(const uint4*)(arow + kb + half * 8 + 16);
    b.q[0] = *(const uint4*)(brow + kb + half * 8);
    b.q[1] = *(const uint4*)(brow + kb + half * 8 + 16);
    acc = wmma_bf16(a, b, acc);
  }
  return acc;
}

// ---------------------------------------------------------------------------
// f32 -> bf16 convert (and transposing variant for W1/W2 so B is stored [N,K])
// ---------------------------------------------------------------------------
__global__ void cvt_bf16_kernel(const float* __restrict__ src,
                                unsigned short* __restrict__ dst, int n) {
  int i = blockIdx.x * blockDim.x + threadIdx.x;
  if (i < n) dst[i] = f2bf(src[i]);
}

__global__ void transpose_cvt_kernel(const float* __restrict__ src,
                                     unsigned short* __restrict__ dst,
                                     int R, int C) {
  int i = blockIdx.x * blockDim.x + threadIdx.x;
  if (i >= R * C) return;
  int rr = i / C, cc = i - rr * C;
  dst[(size_t)cc * R + rr] = f2bf(src[i]);
}

// ---------------------------------------------------------------------------
// Adapter GEMM1: C1[b,s,d] = seq_bf[b,s,:] @ W1[lang_b] + b1[lang_b]  (f32)
// grid.x = B * (H/64);  one block per (batch, 64-col panel).
// ---------------------------------------------------------------------------
__global__ void __launch_bounds__(256) gemm1_kernel(
    const unsigned short* __restrict__ seq_bf,
    const unsigned short* __restrict__ W1T_bf,
    const float* __restrict__ b1,
    const int* __restrict__ lang_ids,
    float* __restrict__ C1) {
  __shared__ __align__(16) unsigned short smem[2 * 64 * BSTRIDE_];
  int bx = blockIdx.x;
  int b = bx >> 4;
  int panelCol0 = (bx & 15) * 64;
  int lang = lang_ids[b];
  const unsigned short* Abase = seq_bf + (size_t)b * S_ * H_;
  const unsigned short* Bt = W1T_bf + (size_t)lang * H_ * H_;
  v8f acc[2][4] = {};
  gemm_panel(Abase, H_, Bt, H_, H_, panelCol0, smem, acc);

  int tid = threadIdx.x, w = tid >> 5, lane = tid & 31;
  int r = lane & 15, half = lane >> 4;
  for (int n = 0; n < 4; ++n) {
    int col = panelCol0 + n * 16 + r;
    float bias = b1[lang * H_ + col];
    for (int tm = 0; tm < 2; ++tm)
      for (int e = 0; e < 8; ++e) {
        int row = w * 32 + tm * 16 + e + half * 8;
        C1[((size_t)(b * S_ + row)) * H_ + col] = acc[tm][n][e] + bias;
      }
  }
}

// ---------------------------------------------------------------------------
// LayerNorm(H=1024) + ReLU + bf16 convert.  One 256-thread block per (b,s).
// ---------------------------------------------------------------------------
__global__ void ln_relu_kernel(const float* __restrict__ C1,
                               const float* __restrict__ ln_g,
                               const float* __restrict__ ln_b,
                               const int* __restrict__ lang_ids,
                               unsigned short* __restrict__ a_bf) {
  int row  = blockIdx.x;               // 0 .. B*S-1
  int b    = row >> 8;                 // S = 256
  int lang = lang_ids[b];
  const float* x = C1 + (size_t)row * H_;
  float v[4], s = 0.f, ss = 0.f;
  for (int i = 0; i < 4; ++i) {
    v[i] = x[threadIdx.x + i * 256];
    s += v[i];
    ss += v[i] * v[i];
  }
  __shared__ float red[256], red2[256];
  red[threadIdx.x] = s;
  red2[threadIdx.x] = ss;
  __syncthreads();
  for (int off = 128; off > 0; off >>= 1) {
    if (threadIdx.x < off) {
      red[threadIdx.x]  += red[threadIdx.x + off];
      red2[threadIdx.x] += red2[threadIdx.x + off];
    }
    __syncthreads();
  }
  float mean = red[0] * (1.0f / H_);
  float var  = red2[0] * (1.0f / H_) - mean * mean;
  float rstd = rsqrtf(var + 1e-5f);
  for (int i = 0; i < 4; ++i) {
    int hc = threadIdx.x + i * 256;
    float y = (v[i] - mean) * rstd * ln_g[lang * H_ + hc] + ln_b[lang * H_ + hc];
    y = fmaxf(y, 0.f);
    a_bf[(size_t)row * H_ + hc] = f2bf(y);
  }
}

// ---------------------------------------------------------------------------
// Adapter GEMM2: adapted_bf[b,s,h] = a_bf[b,s,:] @ W2[lang_b] + b2  (bf16 out)
// ---------------------------------------------------------------------------
__global__ void __launch_bounds__(256) gemm2_kernel(
    const unsigned short* __restrict__ a_bf,
    const unsigned short* __restrict__ W2T_bf,
    const float* __restrict__ b2,
    const int* __restrict__ lang_ids,
    unsigned short* __restrict__ adapted_bf) {
  __shared__ __align__(16) unsigned short smem[2 * 64 * BSTRIDE_];
  int bx = blockIdx.x;
  int b = bx >> 4;
  int panelCol0 = (bx & 15) * 64;
  int lang = lang_ids[b];
  const unsigned short* Abase = a_bf + (size_t)b * S_ * H_;
  const unsigned short* Bt = W2T_bf + (size_t)lang * H_ * H_;
  v8f acc[2][4] = {};
  gemm_panel(Abase, H_, Bt, H_, H_, panelCol0, smem, acc);

  int tid = threadIdx.x, w = tid >> 5, lane = tid & 31;
  int r = lane & 15, half = lane >> 4;
  for (int n = 0; n < 4; ++n) {
    int col = panelCol0 + n * 16 + r;
    float bias = b2[lang * H_ + col];
    for (int tm = 0; tm < 2; ++tm)
      for (int e = 0; e < 8; ++e) {
        int row = w * 32 + tm * 16 + e + half * 8;
        adapted_bf[((size_t)(b * S_ + row)) * H_ + col] = f2bf(acc[tm][n][e] + bias);
      }
  }
}

// ---------------------------------------------------------------------------
// LSTM input precompute: xw[dir][b,s,:] = adapted @ Wih^T + bias  (f32 out)
// Bt = Wih stored as-is [2048,1024].  grid = (32 Mblk * 32 panels, 2 dirs).
// ---------------------------------------------------------------------------
__global__ void __launch_bounds__(256) xw_kernel(
    const unsigned short* __restrict__ adapted_bf,
    const unsigned short* __restrict__ Wih_f_bf,
    const unsigned short* __restrict__ Wih_b_bf,
    const float* __restrict__ bias_f,
    const float* __restrict__ bias_b,
    float* __restrict__ xw_f,
    float* __restrict__ xw_b) {
  __shared__ __align__(16) unsigned short smem[2 * 64 * BSTRIDE_];
  int dir = blockIdx.y;
  const unsigned short* Wih = dir ? Wih_b_bf : Wih_f_bf;
  const float* bias = dir ? bias_b : bias_f;
  float* out = dir ? xw_b : xw_f;
  int bx = blockIdx.x;
  int mblk = bx >> 5;
  int panelCol0 = (bx & 31) * 64;
  const unsigned short* Abase = adapted_bf + (size_t)mblk * 256 * H_;
  v8f acc[2][4] = {};
  gemm_panel(Abase, H_, Wih, H_, H_, panelCol0, smem, acc);

  int tid = threadIdx.x, w = tid >> 5, lane = tid & 31;
  int r = lane & 15, half = lane >> 4;
  for (int n = 0; n < 4; ++n) {
    int col = panelCol0 + n * 16 + r;
    float bv = bias[col];
    for (int tm = 0; tm < 2; ++tm)
      for (int e = 0; e < 8; ++e) {
        int row = mblk * 256 + w * 32 + tm * 16 + e + half * 8;
        out[((size_t)row) * G4_ + col] = acc[tm][n][e] + bv;
      }
  }
}

// ---------------------------------------------------------------------------
// Persistent BiLSTM recurrence.  grid = 2 blocks (one per direction),
// blockDim = 1024 (32 waves).  h kept in LDS as bf16 [32 x 512]; c in regs.
// Wave w owns hidden columns [16w,16w+16) and computes the i/f/g/o gate tiles
// for those columns (gate-col tiles w, w+32, w+64, w+96) for both 16-row
// batch halves, so the pointwise update is wave-local (identical acc layout).
// ---------------------------------------------------------------------------
__global__ void __launch_bounds__(1024) lstm_kernel(
    const float* __restrict__ xw_f, const float* __restrict__ xw_b,
    const unsigned short* __restrict__ Whh_f_bf,
    const unsigned short* __restrict__ Whh_b_bf,
    const int* __restrict__ attn,
    unsigned short* __restrict__ ysbf) {
  const int dir = blockIdx.x;
  const float* xw = dir ? xw_b : xw_f;
  const unsigned short* Whh = dir ? Whh_b_bf : Whh_f_bf;

  __shared__ __align__(16) unsigned short h_bf[B_ * HL_];   // 32 KB

  int tid  = threadIdx.x;
  int w    = tid >> 5;
  int lane = tid & 31;
  int r = lane & 15, half = lane >> 4;
  int ncol = w * 16 + r;                       // hidden unit this lane owns

  for (int i = tid; i < B_ * HL_; i += 1024) h_bf[i] = 0;   // h0 = 0

  float creg[2][8];
  for (int tm = 0; tm < 2; ++tm)
    for (int e = 0; e < 8; ++e) creg[tm][e] = 0.f;          // c0 = 0
  __syncthreads();

  for (int step = 0; step < S_; ++step) {
    int t = dir ? (S_ - 1 - step) : step;

    // init accumulators from precomputed x@Wih^T + bias
    v8f acc[2][4];
    for (int tm = 0; tm < 2; ++tm)
      for (int q = 0; q < 4; ++q) {
        int gcol = q * HL_ + ncol;
        for (int e = 0; e < 8; ++e) {
          int b = tm * 16 + e + half * 8;
          acc[tm][q][e] = xw[((size_t)(b * S_ + t)) * G4_ + gcol];
        }
      }

    // accumulate h @ Whh^T  (A from LDS, Bt from global; K = 512)
    for (int tm = 0; tm < 2; ++tm) {
      const unsigned short* arow = &h_bf[(tm * 16 + r) * HL_];
      for (int q = 0; q < 4; ++q) {
        int jt = q * 32 + w;
        const unsigned short* brow = Whh + ((size_t)(jt * 16 + r)) * HL_;
        acc[tm][q] = wmma_loop(arow, brow, HL_, half, acc[tm][q]);
      }
    }
    __syncthreads();   // all waves done reading h_bf

    // pointwise LSTM update (wave-local: i/f/g/o share the acc coordinates)
    for (int tm = 0; tm < 2; ++tm) {
      for (int e = 0; e < 8; ++e) {
        int b = tm * 16 + e + half * 8;
        bool m = attn[b * S_ + t] != 0;
        float iv = acc[tm][0][e], fv = acc[tm][1][e];
        float gv = acc[tm][2][e], ov = acc[tm][3][e];
        float c_old = creg[tm][e];
        float c_new = sigm_(fv) * c_old + sigm_(iv) * tanhf(gv);
        float h_new = sigm_(ov) * tanhf(c_new);
        creg[tm][e] = m ? c_new : c_old;
        float h_old = bf2f(h_bf[b * HL_ + ncol]);
        h_bf[b * HL_ + ncol] = f2bf(m ? h_new : h_old);
        float y = m ? h_new : 0.f;
        ysbf[((size_t)(b * S_ + t)) * (2 * HL_) + dir * HL_ + ncol] = f2bf(y);
      }
    }
    __syncthreads();   // h_bf updated before next step's WMMA reads
  }
}

// ---------------------------------------------------------------------------
// Projection: out[b,s,:] = ysbf[b,s,:] @ Wp^T + bp   (Bt = Wp as-is [256,1024])
// grid = 32 Mblk * 4 panels.
// ---------------------------------------------------------------------------
__global__ void __launch_bounds__(256) proj_kernel(
    const unsigned short* __restrict__ ysbf,
    const unsigned short* __restrict__ Wp_bf,
    const float* __restrict__ bp,
    float* __restrict__ out) {
  __shared__ __align__(16) unsigned short smem[2 * 64 * BSTRIDE_];
  int bx = blockIdx.x;
  int mblk = bx >> 2;
  int panelCol0 = (bx & 3) * 64;
  const unsigned short* Abase = ysbf + (size_t)mblk * 256 * (2 * HL_);
  v8f acc[2][4] = {};
  gemm_panel(Abase, 2 * HL_, Wp_bf, 2 * HL_, 2 * HL_, panelCol0, smem, acc);

  int tid = threadIdx.x, w = tid >> 5, lane = tid & 31;
  int r = lane & 15, half = lane >> 4;
  for (int n = 0; n < 4; ++n) {
    int col = panelCol0 + n * 16 + r;
    float bv = bp[col];
    for (int tm = 0; tm < 2; ++tm)
      for (int e = 0; e < 8; ++e) {
        int row = mblk * 256 + w * 32 + tm * 16 + e + half * 8;
        out[((size_t)row) * E_ + col] = acc[tm][n][e] + bv;
      }
  }
}

// ---------------------------------------------------------------------------
extern "C" void kernel_launch(void* const* d_in, const int* in_sizes, int n_in,
                              void* d_out, int out_size, void* d_ws, size_t ws_size,
                              hipStream_t stream) {
  (void)in_sizes; (void)n_in; (void)out_size; (void)ws_size;
  const float* seq   = (const float*)d_in[0];
  const int*   attn  = (const int*)d_in[1];
  const int*   lang  = (const int*)d_in[2];
  const float* W1    = (const float*)d_in[3];
  const float* b1    = (const float*)d_in[4];
  const float* ln_g  = (const float*)d_in[5];
  const float* ln_b  = (const float*)d_in[6];
  const float* W2    = (const float*)d_in[7];
  const float* b2    = (const float*)d_in[8];
  const float* Wih_f = (const float*)d_in[9];
  const float* Whh_f = (const float*)d_in[10];
  const float* b_f   = (const float*)d_in[11];
  const float* Wih_b = (const float*)d_in[12];
  const float* Whh_b = (const float*)d_in[13];
  const float* b_b   = (const float*)d_in[14];
  const float* Wp    = (const float*)d_in[15];
  const float* bp    = (const float*)d_in[16];
  float* out = (float*)d_out;

  char* ws = (char*)d_ws;
  size_t off = 0;
  auto alloc = [&](size_t bytes) -> void* {
    void* p = ws + off;
    off = (off + bytes + 255) & ~(size_t)255;
    return p;
  };

  // Region 0 (exactly 64 MB): seq_bf | C1 | a_bf ; reused later as xw_f.
  size_t r0 = off;
  unsigned short* seq_bf = (unsigned short*)alloc((size_t)B_ * S_ * H_ * 2);
  float*          C1     = (float*)alloc((size_t)B_ * S_ * H_ * 4);
  unsigned short* a_bf   = (unsigned short*)alloc((size_t)B_ * S_ * H_ * 2);
  float* xw_f = (float*)(ws + r0);   // aliases region 0 (dead by then)

  unsigned short* W1T_bf  = (unsigned short*)alloc((size_t)L_ * H_ * H_ * 2);
  unsigned short* W2T_bf  = (unsigned short*)alloc((size_t)L_ * H_ * H_ * 2);
  unsigned short* Wih_fbf = (unsigned short*)alloc((size_t)G4_ * H_ * 2);
  unsigned short* Wih_bbf = (unsigned short*)alloc((size_t)G4_ * H_ * 2);
  unsigned short* Whh_fbf = (unsigned short*)alloc((size_t)G4_ * HL_ * 2);
  unsigned short* Whh_bbf = (unsigned short*)alloc((size_t)G4_ * HL_ * 2);
  unsigned short* Wp_bf   = (unsigned short*)alloc((size_t)E_ * H_ * 2);
  unsigned short* adapted_bf = (unsigned short*)alloc((size_t)B_ * S_ * H_ * 2);
  float*          xw_b    = (float*)alloc((size_t)B_ * S_ * G4_ * 4);
  unsigned short* ysbf    = (unsigned short*)alloc((size_t)B_ * S_ * 2 * HL_ * 2);

  // --- precision conversion / weight transposition ---
  {
    int n = B_ * S_ * H_;
    cvt_bf16_kernel<<<(n + 255) / 256, 256, 0, stream>>>(seq, seq_bf, n);
  }
  for (int l = 0; l < L_; ++l) {
    int n = H_ * H_;
    transpose_cvt_kernel<<<(n + 255) / 256, 256, 0, stream>>>(
        W1 + (size_t)l * n, W1T_bf + (size_t)l * n, H_, H_);
    transpose_cvt_kernel<<<(n + 255) / 256, 256, 0, stream>>>(
        W2 + (size_t)l * n, W2T_bf + (size_t)l * n, H_, H_);
  }
  {
    int n = G4_ * H_;
    cvt_bf16_kernel<<<(n + 255) / 256, 256, 0, stream>>>(Wih_f, Wih_fbf, n);
    cvt_bf16_kernel<<<(n + 255) / 256, 256, 0, stream>>>(Wih_b, Wih_bbf, n);
  }
  {
    int n = G4_ * HL_;
    cvt_bf16_kernel<<<(n + 255) / 256, 256, 0, stream>>>(Whh_f, Whh_fbf, n);
    cvt_bf16_kernel<<<(n + 255) / 256, 256, 0, stream>>>(Whh_b, Whh_bbf, n);
  }
  {
    int n = E_ * H_;
    cvt_bf16_kernel<<<(n + 255) / 256, 256, 0, stream>>>(Wp, Wp_bf, n);
  }

  // --- adapter: GEMM1 -> LN+ReLU -> GEMM2 (only the selected language) ---
  gemm1_kernel<<<B_ * 16, 256, 0, stream>>>(seq_bf, W1T_bf, b1, lang, C1);
  ln_relu_kernel<<<B_ * S_, 256, 0, stream>>>(C1, ln_g, ln_b, lang, a_bf);
  gemm2_kernel<<<B_ * 16, 256, 0, stream>>>(a_bf, W2T_bf, b2, lang, adapted_bf);

  // --- LSTM input GEMMs (both directions; overwrites region 0 with xw_f) ---
  xw_kernel<<<dim3(32 * 32, 2), 256, 0, stream>>>(adapted_bf, Wih_fbf, Wih_bbf,
                                                  b_f, b_b, xw_f, xw_b);

  // --- serial BiLSTM recurrence (1 persistent workgroup per direction) ---
  lstm_kernel<<<2, 1024, 0, stream>>>(xw_f, xw_b, Whh_fbf, Whh_bbf, attn, ysbf);

  // --- output projection ---
  proj_kernel<<<1024 / 8, 256, 0, stream>>>(ysbf, Wp_bf, bp, out);
}
